// PointNet2Seg_11123965297221
// MI455X (gfx1250) — matrix operand order repaired and checked by reading, not compiled, not run on hardware
//
#include <hip/hip_runtime.h>
#include <cstdint>
#include <cmath>

typedef __attribute__((ext_vector_type(16))) _Float16 v16h;
typedef __attribute__((ext_vector_type(8)))  _Float16 v8h;
typedef __attribute__((ext_vector_type(8)))  float    v8f;
typedef v16h __attribute__((aligned(16)))    v16h_a16;   // reduced-alignment view

#define BN_EPS 1e-5f
#define LDS_STRIDE 40   // halves; 80B rows: 16B aligned, bank-conflict free

// ---------------------------------------------------------------------------
// Weight conversion: f32 [co][ci] -> f16 [co][ci_pad], zero padded
// ---------------------------------------------------------------------------
__global__ void cvt_w_kernel(const float* __restrict__ w, _Float16* __restrict__ o,
                             int co, int ci, int ci_pad) {
    int total = co * ci_pad;
    for (int e = blockIdx.x * blockDim.x + threadIdx.x; e < total;
         e += gridDim.x * blockDim.x) {
        int r = e / ci_pad, c = e - r * ci_pad;
        o[e] = (c < ci) ? (_Float16)w[r * ci + c] : (_Float16)0.0f;
    }
}

// ---------------------------------------------------------------------------
// Farthest point sampling: one block per batch. dmin kept in LDS.
// ---------------------------------------------------------------------------
__global__ void fps_kernel(const float* __restrict__ xyz, int N, int npoint,
                           int* __restrict__ fidx) {
    int b = blockIdx.x;
    const float* p = xyz + (size_t)b * N * 3;
    __shared__ float dmin[4096];
    __shared__ float rv[256];
    __shared__ int   ri[256];
    __shared__ int   lastSh;
    int tid = threadIdx.x;
    for (int i = tid; i < N; i += 256) dmin[i] = 1e10f;
    if (tid == 0) { fidx[b * npoint] = 0; lastSh = 0; }
    __syncthreads();
    for (int it = 1; it < npoint; it++) {
        int last = lastSh;
        float lx = p[last * 3 + 0], ly = p[last * 3 + 1], lz = p[last * 3 + 2];
        float bv = -1.0f; int bi = 0x7fffffff;
        for (int i = tid; i < N; i += 256) {
            float dx = p[i * 3 + 0] - lx;
            float dy = p[i * 3 + 1] - ly;
            float dz = p[i * 3 + 2] - lz;
            float d  = dx * dx + dy * dy + dz * dz;
            float dm = fminf(dmin[i], d);
            dmin[i] = dm;
            if (dm > bv) { bv = dm; bi = i; }
        }
        rv[tid] = bv; ri[tid] = bi;
        __syncthreads();
        for (int s = 128; s > 0; s >>= 1) {
            if (tid < s) {
                float ov = rv[tid + s]; int oi = ri[tid + s];
                if (ov > rv[tid] || (ov == rv[tid] && oi < ri[tid])) {
                    rv[tid] = ov; ri[tid] = oi;
                }
            }
            __syncthreads();
        }
        if (tid == 0) { fidx[b * npoint + it] = ri[0]; lastSh = ri[0]; }
        __syncthreads();
    }
}

// ---------------------------------------------------------------------------
// Gather sampled coordinates
// ---------------------------------------------------------------------------
__global__ void gather_xyz_kernel(const float* __restrict__ xyz, const int* __restrict__ fidx,
                                  float* __restrict__ out, int N, int npoint) {
    int total = 8 * npoint;
    int gid = blockIdx.x * blockDim.x + threadIdx.x;
    if (gid >= total) return;
    int b = gid / npoint;
    int id = fidx[gid];
    const float* src = xyz + ((size_t)b * N + id) * 3;
    out[gid * 3 + 0] = src[0];
    out[gid * 3 + 1] = src[1];
    out[gid * 3 + 2] = src[2];
}

// ---------------------------------------------------------------------------
// Brute-force kNN: one thread per query; K smallest squared distances.
// ---------------------------------------------------------------------------
template <int K>
__global__ void knn_kernel(const float* __restrict__ q, const float* __restrict__ db,
                           int M, int Ndb, int* __restrict__ outIdx,
                           float* __restrict__ outW) {
    int gid = blockIdx.x * blockDim.x + threadIdx.x;
    if (gid >= 8 * M) return;
    int b = gid / M;
    const float* Q = q + (size_t)gid * 3;
    const float* D = db + (size_t)b * Ndb * 3;
    float qx = Q[0], qy = Q[1], qz = Q[2];
    float bd[K]; int bi[K];
#pragma unroll
    for (int k = 0; k < K; k++) { bd[k] = 1e30f; bi[k] = 0; }
    float worst = 1e30f; int wslot = 0;
    for (int j = 0; j < Ndb; j++) {
        float dx = D[j * 3 + 0] - qx;
        float dy = D[j * 3 + 1] - qy;
        float dz = D[j * 3 + 2] - qz;
        float d  = dx * dx + dy * dy + dz * dz;
        if (d < worst) {
            bd[wslot] = d; bi[wslot] = j;
            worst = bd[0]; wslot = 0;
            for (int k = 1; k < K; k++)
                if (bd[k] > worst) { worst = bd[k]; wslot = k; }
        }
    }
    for (int k = 0; k < K; k++) outIdx[(size_t)gid * K + k] = bi[k];
    if (outW) {
        float w[K], s = 0.0f;
        for (int k = 0; k < K; k++) { w[k] = 1.0f / (bd[k] + 1e-8f); s += w[k]; }
        for (int k = 0; k < K; k++) outW[(size_t)gid * K + k] = w[k] / s;
    }
}

// ---------------------------------------------------------------------------
// SA1 grouping: X[c][col] = xyz[nidx] - new_xyz, rows 3..31 zero. f16 out.
// ---------------------------------------------------------------------------
__global__ void group_sa1_kernel(const float* __restrict__ xyz, const float* __restrict__ nxyz,
                                 const int* __restrict__ nidx, _Float16* __restrict__ X,
                                 int cols) {
    int col = blockIdx.x * blockDim.x + threadIdx.x;
    if (col >= cols) return;
    int pp = col >> 5;
    int b  = pp >> 10;
    int id = nidx[col];
    const float* s = xyz + ((size_t)b * 4096 + id) * 3;
    const float* c = nxyz + (size_t)pp * 3;
#pragma unroll
    for (int k = 0; k < 3; k++)
        X[(size_t)k * cols + col] = (_Float16)(s[k] - c[k]);
#pragma unroll
    for (int k = 3; k < 32; k++)
        X[(size_t)k * cols + col] = (_Float16)0.0f;
}

// ---------------------------------------------------------------------------
// SA2 grouping: X[c][col] = feats[c][b*dbnp + nidx[col]]
// ---------------------------------------------------------------------------
__global__ void group_feats_kernel(const _Float16* __restrict__ f, const int* __restrict__ nidx,
                                   _Float16* __restrict__ X, int rows, int cols,
                                   int np, int dbnp) {
    size_t total = (size_t)rows * cols;
    for (size_t e = blockIdx.x * (size_t)blockDim.x + threadIdx.x; e < total;
         e += (size_t)gridDim.x * blockDim.x) {
        int c = (int)(e / cols);
        int col = (int)(e - (size_t)c * cols);
        int b = col / (np * 32);
        int id = nidx[col];
        X[e] = f[(size_t)c * (8 * dbnp) + b * dbnp + id];
    }
}

// ---------------------------------------------------------------------------
// Fused GEMM + BN + ReLU using WMMA f32_16x16x32_f16.
// Block tile 64(co) x 64(col), 256 threads = 8 waves, 2 WMMA tiles per wave.
// X is staged into LDS TRANSPOSED (Xt[col][k]) so each B fragment is two
// contiguous 16-byte LDS reads. Padded stride 40 halves kills bank conflicts.
// ---------------------------------------------------------------------------
__global__ void gemm_bn_relu_kernel(const _Float16* __restrict__ W,
                                    const _Float16* __restrict__ X,
                                    _Float16* __restrict__ Y,
                                    const float* __restrict__ gam, const float* __restrict__ bet,
                                    const float* __restrict__ btp, const float* __restrict__ mea,
                                    const float* __restrict__ var,
                                    int ci_pad, int cols) {
    __shared__ __align__(16) _Float16 Wl[64 * LDS_STRIDE];
    __shared__ __align__(16) _Float16 Xt[64 * LDS_STRIDE];
    int tid  = threadIdx.x;
    int wave = tid >> 5, lane = tid & 31;
    int wm = wave >> 2, wn = wave & 3;        // wm 0..1, wn 0..3
    int coBase  = blockIdx.y * 64;
    int colBase = blockIdx.x * 64;
    v8f acc0 = {}, acc1 = {};

    // staging coords
    int wr = tid >> 2, wc = (tid & 3) * 8;    // W: 64 rows x 32 k
    int xr = tid >> 3, xc = (tid & 7) * 8;    // X: 32 k x 64 cols

    for (int kk = 0; kk < ci_pad; kk += 32) {
        // ---- stage W (row-major, same layout) ----
        *(v8h*)&Wl[wr * LDS_STRIDE + wc] =
            *(const v8h*)&W[(size_t)(coBase + wr) * ci_pad + kk + wc];
        // ---- stage X transposed: Xt[col][k] ----
        {
            v8h x = *(const v8h*)&X[(size_t)(kk + xr) * cols + colBase + xc];
#pragma unroll
            for (int j = 0; j < 8; j++)
                Xt[(xc + j) * LDS_STRIDE + xr] = x[j];
        }
        // ---- prefetch next K tiles (global_prefetch_b8) ----
        if (kk + 32 < ci_pad) {
            __builtin_prefetch(&W[(size_t)(coBase + wr) * ci_pad + kk + 32 + wc], 0, 1);
            __builtin_prefetch(&X[(size_t)(kk + 32 + xr) * cols + colBase + xc], 0, 1);
        }
        __syncthreads();

        // ---- fragments ----
        int m  = lane & 15;
        int kb = (lane >> 4) * 8;             // A: lanes 0-15 K 0..7/16..23, 16-31 K 8..15/24..31
        v8h a0lo = *(const v8h*)&Wl[(wm * 16 + m) * LDS_STRIDE + kb];
        v8h a0hi = *(const v8h*)&Wl[(wm * 16 + m) * LDS_STRIDE + 16 + kb];
        v8h a1lo = *(const v8h*)&Wl[((wm + 2) * 16 + m) * LDS_STRIDE + kb];
        v8h a1hi = *(const v8h*)&Wl[((wm + 2) * 16 + m) * LDS_STRIDE + 16 + kb];
        v16h a0, a1;
#pragma unroll
        for (int i = 0; i < 8; i++) {
            a0[i] = a0lo[i]; a0[8 + i] = a0hi[i];
            a1[i] = a1lo[i]; a1[8 + i] = a1hi[i];
        }
        // B: lane n = wn*16 + (lane&15); lanes 0-15 K 0..7/16..23, lanes 16-31 K 8..15/24..31
        int n  = wn * 16 + (lane & 15);
        v8h blo = *(const v8h*)&Xt[n * LDS_STRIDE + kb];
        v8h bhi = *(const v8h*)&Xt[n * LDS_STRIDE + 16 + kb];
        v16h bfr;
#pragma unroll
        for (int i = 0; i < 8; i++) { bfr[i] = blo[i]; bfr[8 + i] = bhi[i]; }

        acc0 = __builtin_amdgcn_wmma_f32_16x16x32_f16(false, a0, false, bfr,
                                                      (short)0, acc0, false, false);
        acc1 = __builtin_amdgcn_wmma_f32_16x16x32_f16(false, a1, false, bfr,
                                                      (short)0, acc1, false, false);
        __syncthreads();
    }

    // ---- epilogue: BN + ReLU, write f16 (two tiles per wave) ----
    int col  = colBase + wn * 16 + (lane & 15);
    int moff = (lane >> 4) * 8;
#pragma unroll
    for (int r = 0; r < 8; r++) {
        int m0 = coBase + wm * 16 + moff + r;
        int m1 = coBase + (wm + 2) * 16 + moff + r;
        float s0 = gam[m0] / sqrtf(var[m0] + BN_EPS);
        float s1 = gam[m1] / sqrtf(var[m1] + BN_EPS);
        float v0 = acc0[r] * s0 + (bet[m0] - mea[m0]) * s0 + btp[m0];
        float v1 = acc1[r] * s1 + (bet[m1] - mea[m1]) * s1 + btp[m1];
        Y[(size_t)m0 * cols + col] = (_Float16)fmaxf(v0, 0.0f);
        Y[(size_t)m1 * cols + col] = (_Float16)fmaxf(v1, 0.0f);
    }
}

// ---------------------------------------------------------------------------
// Max pool over nsample=32
// ---------------------------------------------------------------------------
__global__ void maxpool_kernel(const _Float16* __restrict__ Y, _Float16* __restrict__ f,
                               int co, int npts) {
    size_t total = (size_t)co * npts;
    for (size_t e = blockIdx.x * (size_t)blockDim.x + threadIdx.x; e < total;
         e += (size_t)gridDim.x * blockDim.x) {
        int c  = (int)(e / npts);
        int pt = (int)(e - (size_t)c * npts);
        const _Float16* src = Y + (size_t)c * npts * 32 + (size_t)pt * 32;
        float m = -1e30f;
#pragma unroll
        for (int s = 0; s < 32; s++) m = fmaxf(m, (float)src[s]);
        f[e] = (_Float16)m;
    }
}

// ---------------------------------------------------------------------------
// FP1 input build: rows 0..255 copy f1; rows 256..767 3-NN interp from f2.
// ---------------------------------------------------------------------------
__global__ void build_xfp1_kernel(const _Float16* __restrict__ f1, const _Float16* __restrict__ f2,
                                  const int* __restrict__ idx, const float* __restrict__ w,
                                  _Float16* __restrict__ X) {
    const int cols = 8192;
    size_t total = (size_t)768 * cols;
    for (size_t e = blockIdx.x * (size_t)blockDim.x + threadIdx.x; e < total;
         e += (size_t)gridDim.x * blockDim.x) {
        int row = (int)(e / cols);
        int col = (int)(e - (size_t)row * cols);
        if (row < 256) {
            X[e] = f1[(size_t)row * cols + col];
        } else {
            int c = row - 256;
            int b = col >> 10;
            float acc = 0.0f;
#pragma unroll
            for (int k = 0; k < 3; k++) {
                int id = idx[col * 3 + k];
                acc += w[col * 3 + k] * (float)f2[(size_t)c * 2048 + b * 256 + id];
            }
            X[e] = (_Float16)acc;
        }
    }
}

// ---------------------------------------------------------------------------
// FP2 input build: 3-NN interp of fpo1 (256ch, 8192 db cols) to 32768 cols
// ---------------------------------------------------------------------------
__global__ void build_interp2_kernel(const _Float16* __restrict__ src, const int* __restrict__ idx,
                                     const float* __restrict__ w, _Float16* __restrict__ X) {
    const int cols = 32768;
    size_t total = (size_t)256 * cols;
    for (size_t e = blockIdx.x * (size_t)blockDim.x + threadIdx.x; e < total;
         e += (size_t)gridDim.x * blockDim.x) {
        int c   = (int)(e / cols);
        int col = (int)(e - (size_t)c * cols);
        int b = col >> 12;
        float acc = 0.0f;
#pragma unroll
        for (int k = 0; k < 3; k++) {
            int id = idx[col * 3 + k];
            acc += w[col * 3 + k] * (float)src[(size_t)c * 8192 + b * 1024 + id];
        }
        X[e] = (_Float16)acc;
    }
}

// ---------------------------------------------------------------------------
// Head2: out[col*26+o] = w2[o][:] . h[:, col] + b2[o]   (f32 output)
// ---------------------------------------------------------------------------
__global__ void head2_kernel(const _Float16* __restrict__ h, const float* __restrict__ w2,
                             const float* __restrict__ b2, float* __restrict__ out) {
    const int cols = 32768;
    size_t total = (size_t)cols * 26;
    for (size_t e = blockIdx.x * (size_t)blockDim.x + threadIdx.x; e < total;
         e += (size_t)gridDim.x * blockDim.x) {
        int col = (int)(e / 26);
        int o   = (int)(e - (size_t)col * 26);
        float acc = b2[o];
        for (int ci = 0; ci < 128; ci++)
            acc += w2[o * 128 + ci] * (float)h[(size_t)ci * cols + col];
        out[e] = acc;
    }
}

// ---------------------------------------------------------------------------
// Host side
// ---------------------------------------------------------------------------
extern "C" void kernel_launch(void* const* d_in, const int* in_sizes, int n_in,
                              void* d_out, int out_size, void* d_ws, size_t ws_size,
                              hipStream_t stream) {
    (void)in_sizes; (void)n_in; (void)out_size; (void)ws_size;
    const float* pts = (const float*)d_in[0];

    struct Lyr { const float* p[6]; int ci, co, ci_pad; };
    static const int cis[11] = { 3, 64, 128, 256, 256, 512, 768, 256, 256, 256, 128 };
    static const int cos_[11] = { 64, 128, 256, 256, 512, 512, 256, 256, 256, 128, 128 };
    Lyr ls[11];
    int ii = 1;
    for (int l = 0; l < 11; l++) {
        for (int j = 0; j < 6; j++) ls[l].p[j] = (const float*)d_in[ii++];
        ls[l].ci = cis[l]; ls[l].co = cos_[l];
        ls[l].ci_pad = (cis[l] + 31) / 32 * 32;
    }
    const float* w2h = (const float*)d_in[ii++];
    const float* b2h = (const float*)d_in[ii++];
    float* out = (float*)d_out;

    // --- workspace carving ---
    char* base = (char*)d_ws;
    size_t cur = 0;
    auto alloc = [&](size_t bytes) -> void* {
        cur = (cur + 255) & ~(size_t)255;
        void* r = base + cur;
        cur += bytes;
        return r;
    };
    const size_t cols1 = (size_t)8 * 1024 * 32;   // 262144
    const size_t cols2 = (size_t)8 * 256 * 32;    // 65536

    _Float16* wf[11];
    for (int l = 0; l < 11; l++)
        wf[l] = (_Float16*)alloc((size_t)ls[l].co * ls[l].ci_pad * 2);

    _Float16* Abuf = (_Float16*)alloc((size_t)128 * cols1 * 2);
    _Float16* Bbuf = (_Float16*)alloc((size_t)256 * cols1 * 2);
    _Float16* f1   = (_Float16*)alloc((size_t)256 * 8192 * 2);
    _Float16* f2   = (_Float16*)alloc((size_t)512 * 2048 * 2);
    _Float16* Xfp1 = (_Float16*)alloc((size_t)768 * 8192 * 2);
    _Float16* Yfp1 = (_Float16*)alloc((size_t)256 * 8192 * 2);
    _Float16* fpo1 = (_Float16*)alloc((size_t)256 * 8192 * 2);
    _Float16* Xfp2 = (_Float16*)alloc((size_t)256 * 32768 * 2);
    _Float16* Yfp2 = (_Float16*)alloc((size_t)256 * 32768 * 2);
    _Float16* fpo2 = (_Float16*)alloc((size_t)128 * 32768 * 2);
    _Float16* h1   = (_Float16*)alloc((size_t)128 * 32768 * 2);
    float* nxyz1 = (float*)alloc((size_t)8 * 1024 * 3 * 4);
    float* nxyz2 = (float*)alloc((size_t)8 * 256 * 3 * 4);
    float* wq1   = (float*)alloc((size_t)8 * 1024 * 3 * 4);
    float* wq2   = (float*)alloc((size_t)8 * 4096 * 3 * 4);
    int* fidx1 = (int*)alloc((size_t)8 * 1024 * 4);
    int* fidx2 = (int*)alloc((size_t)8 * 256 * 4);
    int* nidx1 = (int*)alloc(cols1 * 4);
    int* nidx2 = (int*)alloc(cols2 * 4);
    int* idxf1 = (int*)alloc((size_t)8 * 1024 * 3 * 4);
    int* idxf2 = (int*)alloc((size_t)8 * 4096 * 3 * 4);

    // --- weight conversion ---
    for (int l = 0; l < 11; l++) {
        int n = ls[l].co * ls[l].ci_pad;
        cvt_w_kernel<<<(n + 255) / 256, 256, 0, stream>>>(ls[l].p[0], wf[l],
                                                          ls[l].co, ls[l].ci, ls[l].ci_pad);
    }

    auto gemm = [&](int l, const _Float16* X, _Float16* Y, int cols) {
        dim3 grid(cols / 64, ls[l].co / 64);
        gemm_bn_relu_kernel<<<grid, 256, 0, stream>>>(
            wf[l], X, Y, ls[l].p[2], ls[l].p[1], ls[l].p[3], ls[l].p[4], ls[l].p[5],
            ls[l].ci_pad, cols);
    };

    // ---- SA1 ----
    fps_kernel<<<8, 256, 0, stream>>>(pts, 4096, 1024, fidx1);
    gather_xyz_kernel<<<(8 * 1024 + 255) / 256, 256, 0, stream>>>(pts, fidx1, nxyz1, 4096, 1024);
    knn_kernel<32><<<(8 * 1024 + 255) / 256, 256, 0, stream>>>(nxyz1, pts, 1024, 4096, nidx1, nullptr);
    group_sa1_kernel<<<(int)(cols1 / 256), 256, 0, stream>>>(pts, nxyz1, nidx1, Abuf, (int)cols1);
    gemm(0, Abuf, Bbuf, (int)cols1);   //   3->64
    gemm(1, Bbuf, Abuf, (int)cols1);   //  64->128
    gemm(2, Abuf, Bbuf, (int)cols1);   // 128->256
    maxpool_kernel<<<2048, 256, 0, stream>>>(Bbuf, f1, 256, 8192);

    // ---- SA2 ----
    fps_kernel<<<8, 256, 0, stream>>>(nxyz1, 1024, 256, fidx2);
    gather_xyz_kernel<<<(8 * 256 + 255) / 256, 256, 0, stream>>>(nxyz1, fidx2, nxyz2, 1024, 256);
    knn_kernel<32><<<(8 * 256 + 255) / 256, 256, 0, stream>>>(nxyz2, nxyz1, 256, 1024, nidx2, nullptr);
    group_feats_kernel<<<4096, 256, 0, stream>>>(f1, nidx2, Abuf, 256, (int)cols2, 256, 1024);
    gemm(3, Abuf, Bbuf, (int)cols2);   // 256->256
    gemm(4, Bbuf, Abuf, (int)cols2);   // 256->512
    gemm(5, Abuf, Bbuf, (int)cols2);   // 512->512
    maxpool_kernel<<<1024, 256, 0, stream>>>(Bbuf, f2, 512, 2048);

    // ---- FP1 ----
    knn_kernel<3><<<(8 * 1024 + 255) / 256, 256, 0, stream>>>(nxyz1, nxyz2, 1024, 256, idxf1, wq1);
    build_xfp1_kernel<<<4096, 256, 0, stream>>>(f1, f2, idxf1, wq1, Xfp1);
    gemm(6, Xfp1, Yfp1, 8192);         // 768->256
    gemm(7, Yfp1, fpo1, 8192);         // 256->256

    // ---- FP2 ----
    knn_kernel<3><<<(8 * 4096 + 255) / 256, 256, 0, stream>>>(pts, nxyz1, 4096, 1024, idxf2, wq2);
    build_interp2_kernel<<<4096, 256, 0, stream>>>(fpo1, idxf2, wq2, Xfp2);
    gemm(8, Xfp2, Yfp2, 32768);        // 256->256
    gemm(9, Yfp2, fpo2, 32768);        // 256->128

    // ---- Heads ----
    gemm(10, fpo2, h1, 32768);         // 128->128
    head2_kernel<<<2048, 256, 0, stream>>>(h1, w2h, b2h, out);
}
